// GVPModel_38912403702065
// MI455X (gfx1250) — matrix operand order repaired and checked by reading, not compiled, not run on hardware
//
#include <hip/hip_runtime.h>
#include <hip/hip_bf16.h>

// ---------------------------------------------------------------------------
// GVP graph network forward for MI455X (gfx1250, wave32, WMMA).
// Dominant compute (edge message GEMMs, node FF GEMMs) runs through
// v_wmma_f32_16x16x32_f16 with 128-bit fragment loads (A row-major,
// B packed transposed so both fragments are contiguous); everything else is
// scalar VALU / atomics. Workspace use ~560 MB.
// ---------------------------------------------------------------------------

#define NN 10000      // N_NODES (multiple of 16)
#define NE 320000     // N_EDGES (multiple of 16)
#define NG 32         // N_GRAPHS
#define LDN 112       // padded leading dim for 100-wide scalar features

typedef _Float16 f16;
typedef __attribute__((ext_vector_type(16))) _Float16 v16h;
typedef __attribute__((ext_vector_type(8)))  _Float16 v8h;
typedef __attribute__((ext_vector_type(8)))  float    v8f;

__device__ __forceinline__ float sigf(float x){ return 1.f/(1.f+__expf(-x)); }
__device__ __forceinline__ float reluf(float x){ return x>0.f?x:0.f; }
__device__ __forceinline__ float lreluf(float x){ return x>0.f?x:0.1f*x; }

__device__ __forceinline__ v16h frag_cat(v8h lo, v8h hi)
{
    v16h r;
#pragma unroll
    for (int e = 0; e < 8; ++e) { r[e] = lo[e]; r[e+8] = hi[e]; }
    return r;
}

// ---------------------------------------------------------------------------
// WMMA GEMM: C[M,Np] = act(A_f16[M,Kp] @ B[Kp,Np] + biasPad)
//   A: row-major f16, lda = Kp (multiple of 32, 64B-aligned rows)
//   Bt: weight packed TRANSPOSED f16 [Np,Kp] so each lane's B fragment is a
//       pair of contiguous 8xf16 runs -> 128-bit loads
//   biasPad: f32[Np], zero-padded
// grid = (M/16, Np/16), block = 32 (one wave per 16x16 tile; EXEC all ones).
// Fragment layouts per CDNA5 ISA 7.12.2:
//   A 16x32: lane M = lane%16; lane<16 K {0..7,16..23}; lane>=16 K {8..15,24..31}
//   B 32x16: lane N = lane%16; lane<16 K {0..7,16..23}; lane>=16 K {8..15,24..31}
//   C 16x16: VGPR r, lane -> M = r + 8*(lane>=16), N = lane%16
// ---------------------------------------------------------------------------
__global__ void k_gemm_wmma(const f16* __restrict__ A, int lda,
                            const f16* __restrict__ Bt, int ldb,
                            const float* __restrict__ biasPad,
                            float* __restrict__ C, int ldc,
                            int kTiles, int act)
{
    const int mt = blockIdx.x, nt = blockIdx.y;
    const int lane   = threadIdx.x & 31;
    const int laneHi = lane >> 4;
    const int laneLo = lane & 15;
    const f16* Arow = A  + (size_t)(mt*16 + laneLo) * lda + 8*laneHi;
    const f16* Brow = Bt + (size_t)(nt*16 + laneLo) * ldb + 8*laneHi;
    v8f acc = {};

#define KSTEP(K0)                                                              \
    {                                                                          \
        v16h a = frag_cat(*(const v8h*)(Arow + (K0)),                          \
                          *(const v8h*)(Arow + (K0) + 16));                    \
        v16h b = frag_cat(*(const v8h*)(Brow + (K0)),                          \
                          *(const v8h*)(Brow + (K0) + 16));                    \
        acc = __builtin_amdgcn_wmma_f32_16x16x32_f16(                          \
                  false, a, false, b, (short)0, acc, false, false);            \
    }

    int kt = 0;
    for (; kt + 2 <= kTiles; kt += 2) {
        // gfx1250 global_prefetch_b8 of the A stream two tiles ahead
        __builtin_prefetch((const void*)(Arow + kt*32 + 64), 0, 1);
        KSTEP(kt*32);
        KSTEP(kt*32 + 32);
    }
    if (kt < kTiles) KSTEP(kt*32);
#undef KSTEP

    const float bv = biasPad[nt*16 + laneLo];
#pragma unroll
    for (int r = 0; r < 8; ++r) {
        const int row = mt*16 + r + 8*laneHi;
        const int col = nt*16 + laneLo;
        float x = acc[r] + bv;
        if (act == 1) x = reluf(x);
        C[(size_t)row * ldc + col] = x;
    }
}

// Pack f32 weight [K,N] into zero-padded TRANSPOSED f16 [Np,Kp]
__global__ void k_pack_wT(const float* __restrict__ W, int K, int N,
                          f16* __restrict__ Bt, int Kp, int Np)
{
    int idx = blockIdx.x*blockDim.x + threadIdx.x;
    if (idx >= Kp*Np) return;
    int n = idx / Kp, k = idx % Kp;
    Bt[idx] = (k < K && n < N) ? (f16)W[(size_t)k*N + n] : (f16)0.f;
}

__global__ void k_pack_bias(const float* __restrict__ b, int N,
                            float* __restrict__ bp, int Np)
{
    int i = blockIdx.x*blockDim.x + threadIdx.x;
    if (i < Np) bp[i] = (i < N) ? b[i] : 0.f;
}

// ---------------------------------------------------------------------------
// layer_norm_tuple with optional residual: s' = LN(s+ds)*g+b ; V' = (V+dV)/vn
// ---------------------------------------------------------------------------
__global__ void k_ln_res(const float* __restrict__ sIn, int ldSi,
                         const float* __restrict__ dS, int ldD, int ns,
                         const float* __restrict__ Vin, const float* __restrict__ dV, int nv,
                         const float* __restrict__ g, const float* __restrict__ b,
                         float* __restrict__ sOut, int ldSo,
                         float* __restrict__ Vout, int rows)
{
    int r = blockIdx.x*blockDim.x + threadIdx.x;
    if (r >= rows) return;
    float mu = 0.f;
    for (int i = 0; i < ns; ++i) {
        float x = sIn[(size_t)r*ldSi + i];
        if (dS) x += dS[(size_t)r*ldD + i];
        mu += x;
    }
    mu /= ns;
    float var = 0.f;
    for (int i = 0; i < ns; ++i) {
        float x = sIn[(size_t)r*ldSi + i];
        if (dS) x += dS[(size_t)r*ldD + i];
        float d = x - mu; var += d*d;
    }
    var /= ns;
    float inv = rsqrtf(var + 1e-5f);
    for (int i = 0; i < ns; ++i) {
        float x = sIn[(size_t)r*ldSi + i];
        if (dS) x += dS[(size_t)r*ldD + i];
        sOut[(size_t)r*ldSo + i] = (x - mu)*inv*g[i] + b[i];
    }
    if (nv > 0) {
        float acc = 0.f;
        for (int v = 0; v < nv; ++v) {
            float ss = 0.f;
            for (int c = 0; c < 3; ++c) {
                float x = Vin[((size_t)r*nv + v)*3 + c];
                if (dV) x += dV[((size_t)r*nv + v)*3 + c];
                ss += x*x;
            }
            acc += fmaxf(ss, 1e-8f);
        }
        float invv = rsqrtf(acc / nv);
        for (int v = 0; v < nv; ++v)
            for (int c = 0; c < 3; ++c) {
                float x = Vin[((size_t)r*nv + v)*3 + c];
                if (dV) x += dV[((size_t)r*nv + v)*3 + c];
                Vout[((size_t)r*nv + v)*3 + c] = x*invv;
            }
    }
}

// ---------------------------------------------------------------------------
// Generic GVP vector track + f16 A-matrix staging:
//   Vh = Vin@wh (per xyz comp); A row = [sIn | ||Vh_h|| | 0-pad]; Vo = Vh@wv (gated)
// ---------------------------------------------------------------------------
__global__ void k_gvp_vec(const float* __restrict__ sIn, int ldS, int si,
                          const float* __restrict__ Vin, int vi,
                          const float* __restrict__ wh, int h,
                          const float* __restrict__ wv, int vo, int gateFlag,
                          float* __restrict__ Vout,
                          f16* __restrict__ A, int Kp, int rows)
{
    int r = blockIdx.x*blockDim.x + threadIdx.x;
    if (r >= rows) return;
    f16* a = A + (size_t)r * Kp;
    for (int c = 0; c < si; ++c) a[c] = (f16)sIn[(size_t)r*ldS + c];
    float Vm[33][3];
    for (int v = 0; v < vi; ++v)
        for (int c = 0; c < 3; ++c) Vm[v][c] = Vin[((size_t)r*vi + v)*3 + c];
    float Vo[32][3];
    for (int v = 0; v < vo; ++v) { Vo[v][0]=0.f; Vo[v][1]=0.f; Vo[v][2]=0.f; }
    for (int hh = 0; hh < h; ++hh) {
        float vh0=0.f, vh1=0.f, vh2=0.f;
        for (int v = 0; v < vi; ++v) {
            float w = wh[(size_t)v*h + hh];
            vh0 += Vm[v][0]*w; vh1 += Vm[v][1]*w; vh2 += Vm[v][2]*w;
        }
        float ss = vh0*vh0 + vh1*vh1 + vh2*vh2;
        a[si + hh] = (f16)sqrtf(fmaxf(ss, 1e-8f));
        for (int v = 0; v < vo; ++v) {
            float w = wv[(size_t)hh*vo + v];
            Vo[v][0] += vh0*w; Vo[v][1] += vh1*w; Vo[v][2] += vh2*w;
        }
    }
    for (int c = si + h; c < Kp; ++c) a[c] = (f16)0.f;
    for (int v = 0; v < vo; ++v) {
        float sc = 1.f;
        if (gateFlag) {
            float ss = Vo[v][0]*Vo[v][0] + Vo[v][1]*Vo[v][1] + Vo[v][2]*Vo[v][2];
            sc = sigf(sqrtf(fmaxf(ss, 1e-8f)));
        }
        for (int c = 0; c < 3; ++c)
            Vout[((size_t)r*vo + v)*3 + c] = Vo[v][c]*sc;
    }
}

// ---------------------------------------------------------------------------
// Edge message stage 0: gather concat [s_src|e_s|s_dst|vn] and vector GVP
// (vi=33 = 16(src)+1(edge)+16(dst), h=33, vo=16, sigmoid gate). Kp=288.
// ---------------------------------------------------------------------------
__global__ void k_edge_msg0(const float* __restrict__ sN, int ldS,
                            const float* __restrict__ VN,
                            const float* __restrict__ esE, int ldE,
                            const float* __restrict__ eVE,
                            const int* __restrict__ src, const int* __restrict__ dst,
                            const float* __restrict__ wh, const float* __restrict__ wv,
                            float* __restrict__ VoE, f16* __restrict__ A, int Kp)
{
    int e = blockIdx.x*blockDim.x + threadIdx.x;
    if (e >= NE) return;
    const int j = src[e], i = dst[e];
    f16* a = A + (size_t)e * Kp;
    for (int c = 0; c < 100; ++c) a[c]       = (f16)sN[(size_t)j*ldS + c];
    for (int c = 0; c < 32;  ++c) a[100 + c] = (f16)esE[(size_t)e*ldE + c];
    for (int c = 0; c < 100; ++c) a[132 + c] = (f16)sN[(size_t)i*ldS + c];
    float Vm[33][3];
    for (int v = 0; v < 16; ++v)
        for (int c = 0; c < 3; ++c) Vm[v][c] = VN[((size_t)j*16 + v)*3 + c];
    for (int c = 0; c < 3; ++c) Vm[16][c] = eVE[(size_t)e*3 + c];
    for (int v = 0; v < 16; ++v)
        for (int c = 0; c < 3; ++c) Vm[17+v][c] = VN[((size_t)i*16 + v)*3 + c];
    float Vo[16][3];
    for (int v = 0; v < 16; ++v) { Vo[v][0]=0.f; Vo[v][1]=0.f; Vo[v][2]=0.f; }
    for (int hh = 0; hh < 33; ++hh) {
        float vh0=0.f, vh1=0.f, vh2=0.f;
        for (int v = 0; v < 33; ++v) {
            float w = wh[v*33 + hh];
            vh0 += Vm[v][0]*w; vh1 += Vm[v][1]*w; vh2 += Vm[v][2]*w;
        }
        float ss = vh0*vh0 + vh1*vh1 + vh2*vh2;
        a[232 + hh] = (f16)sqrtf(fmaxf(ss, 1e-8f));
        for (int v = 0; v < 16; ++v) {
            float w = wv[hh*16 + v];
            Vo[v][0] += vh0*w; Vo[v][1] += vh1*w; Vo[v][2] += vh2*w;
        }
    }
    for (int c = 265; c < Kp; ++c) a[c] = (f16)0.f;
    for (int v = 0; v < 16; ++v) {
        float ss = Vo[v][0]*Vo[v][0] + Vo[v][1]*Vo[v][1] + Vo[v][2]*Vo[v][2];
        float sc = sigf(sqrtf(fmaxf(ss, 1e-8f)));
        for (int c = 0; c < 3; ++c) VoE[((size_t)e*16 + v)*3 + c] = Vo[v][c]*sc;
    }
}

// --------------------------- small utility kernels --------------------------
__global__ void k_zero(float* p, long n)
{
    for (long i = blockIdx.x*(long)blockDim.x + threadIdx.x; i < n;
         i += (long)gridDim.x*blockDim.x) p[i] = 0.f;
}

__global__ void k_count(const int* __restrict__ idx, int n, float* __restrict__ cnt)
{
    int i = blockIdx.x*blockDim.x + threadIdx.x;
    if (i < n) atomicAdd(&cnt[idx[i]], 1.f);
}

__global__ void k_scatter_add(const float* __restrict__ X, int ldX, int cols,
                              const int* __restrict__ idx, int rows,
                              float* __restrict__ out, int ldO)
{
    long t = blockIdx.x*(long)blockDim.x + threadIdx.x;
    if (t >= (long)rows*cols) return;
    int r = (int)(t / cols), c = (int)(t % cols);
    atomicAdd(&out[(size_t)idx[r]*ldO + c], X[(size_t)r*ldX + c]);
}

__global__ void k_div_cnt(float* __restrict__ X, int ld, int cols,
                          const float* __restrict__ cnt, int rows)
{
    long t = blockIdx.x*(long)blockDim.x + threadIdx.x;
    if (t >= (long)rows*cols) return;
    int r = (int)(t / cols), c = (int)(t % cols);
    X[(size_t)r*ld + c] /= fmaxf(cnt[r], 1.f);
}

__global__ void k_apply_mask(float* __restrict__ sNew, int ldN,
                             const float* __restrict__ sOld, int ldO,
                             float* __restrict__ Vnew, const float* __restrict__ Vold,
                             const int* __restrict__ wsh, int rows)
{
    int r = blockIdx.x*blockDim.x + threadIdx.x;
    if (r >= rows || wsh[r] > 0) return;
    for (int c = 0; c < 100; ++c) sNew[(size_t)r*ldN + c] = sOld[(size_t)r*ldO + c];
    for (int c = 0; c < 48;  ++c) Vnew[(size_t)r*48  + c] = Vold[(size_t)r*48  + c];
}

__global__ void k_add_gc(float* __restrict__ out, int ldO,
                         const float* __restrict__ gc, int ldG,
                         const int* __restrict__ batch, int rows)
{
    long t = blockIdx.x*(long)blockDim.x + threadIdx.x;
    if (t >= (long)rows*100) return;
    int r = (int)(t / 100), c = (int)(t % 100);
    out[(size_t)r*ldO + c] += 0.5f*gc[(size_t)batch[r]*ldG + c];
}

__global__ void k_gate(const float* __restrict__ out, int ldO,
                       const float* __restrict__ w1, const float* __restrict__ b1,
                       const float* __restrict__ w2, const float* __restrict__ b2,
                       float* __restrict__ g, int rows)
{
    int n = blockIdx.x*blockDim.x + threadIdx.x;
    if (n >= rows) return;
    float acc = b2[0];
    for (int hh = 0; hh < 16; ++hh) {
        float a = b1[hh];
        for (int c = 0; c < 100; ++c) a += out[(size_t)n*ldO + c]*w1[c*16 + hh];
        acc += lreluf(a)*w2[hh];
    }
    g[n] = acc;
}

// GlobalAttention pool: one block per graph
__global__ void k_pool(const float* __restrict__ out, int ldO,
                       const float* __restrict__ gv, const int* __restrict__ batch,
                       float* __restrict__ rep, int ldR)
{
    const int g = blockIdx.x, tid = threadIdx.x;
    __shared__ float red[256];
    __shared__ float repSh[112];
    float m = -1e30f;
    for (int n = tid; n < NN; n += 256) if (batch[n] == g) m = fmaxf(m, gv[n]);
    red[tid] = m; __syncthreads();
    for (int s = 128; s > 0; s >>= 1) { if (tid < s) red[tid] = fmaxf(red[tid], red[tid+s]); __syncthreads(); }
    float gmax = red[0]; if (!(gmax > -1e29f)) gmax = 0.f;
    __syncthreads();
    float se = 0.f;
    for (int n = tid; n < NN; n += 256) if (batch[n] == g) se += __expf(gv[n] - gmax);
    red[tid] = se; __syncthreads();
    for (int s = 128; s > 0; s >>= 1) { if (tid < s) red[tid] += red[tid+s]; __syncthreads(); }
    float denom = fmaxf(red[0], 1e-20f);
    if (tid < 112) repSh[tid] = 0.f;
    __syncthreads();
    for (int n = tid; n < NN; n += 256) if (batch[n] == g) {
        float alpha = __expf(gv[n] - gmax) / denom;
        for (int c = 0; c < 100; ++c) atomicAdd(&repSh[c], alpha*out[(size_t)n*ldO + c]);
    }
    __syncthreads();
    if (tid < 100) rep[(size_t)g*ldR + tid] = repSh[tid];
}

// Final MLPWithResidual head: 32 graphs, one thread each
__global__ void k_mlp(const float* __restrict__ rep, int ldR,
                      const float* w0, const float* b0,
                      const float* r0w1, const float* r0b1,
                      const float* r0w2, const float* r0b2,
                      const float* w1, const float* b1,
                      const float* r1w1, const float* r1b1,
                      const float* r1w2, const float* r1b2,
                      const float* w2, const float* b2,
                      float* __restrict__ outp)
{
    int g = threadIdx.x;
    if (g >= NG) return;
    float x[64], h1[64], h2[64], y[32], z1[32], z2[32];
    for (int o = 0; o < 64; ++o) {
        float a = b0[o];
        for (int c = 0; c < 100; ++c) a += rep[(size_t)g*ldR + c]*w0[c*64 + o];
        x[o] = lreluf(a);
    }
    for (int o = 0; o < 64; ++o) { float a = r0b1[o]; for (int c = 0; c < 64; ++c) a += x[c]*r0w1[c*64+o]; h1[o] = reluf(a); }
    for (int o = 0; o < 64; ++o) { float a = r0b2[o]; for (int c = 0; c < 64; ++c) a += h1[c]*r0w2[c*64+o]; h2[o] = reluf(a); }
    for (int o = 0; o < 64; ++o) x[o] = h2[o] + x[o];
    for (int o = 0; o < 32; ++o) { float a = b1[o]; for (int c = 0; c < 64; ++c) a += x[c]*w1[c*32+o]; y[o] = lreluf(a); }
    for (int o = 0; o < 32; ++o) { float a = r1b1[o]; for (int c = 0; c < 32; ++c) a += y[c]*r1w1[c*32+o]; z1[o] = reluf(a); }
    for (int o = 0; o < 32; ++o) { float a = r1b2[o]; for (int c = 0; c < 32; ++c) a += z1[c]*r1w2[c*32+o]; z2[o] = reluf(a); }
    for (int o = 0; o < 32; ++o) y[o] = z2[o] + y[o];
    float a = b2[0];
    for (int c = 0; c < 32; ++c) a += y[c]*w2[c];
    outp[g] = a;
}

// ---------------------------------------------------------------------------
// Host orchestration
// ---------------------------------------------------------------------------
struct GvpW { const float *wh, *ws_b, *ws_w, *wv; };

extern "C" void kernel_launch(void* const* d_in, const int* in_sizes, int n_in,
                              void* d_out, int out_size, void* d_ws, size_t ws_size,
                              hipStream_t stream)
{
    (void)in_sizes; (void)n_in; (void)out_size; (void)ws_size;
    // ---- params in JAX tree-flatten (alphabetical dict key) order ----
    int ci = 0;
    auto F = [&]() { return (const float*)d_in[ci++]; };
    // 'We': ln_b, ln_g, wh(1,1), ws_b(32), ws_w(33,32), wv(1,1)
    const float* We_ln_b = F(); const float* We_ln_g = F();
    GvpW We; We.wh = F(); We.ws_b = F(); We.ws_w = F(); We.wv = F();
    // 'Wout': ln_b, ln_g, wh(16,16), ws_b(100), ws_w(116,100)
    const float* Wo_ln_b = F(); const float* Wo_ln_g = F();
    GvpW Wo; Wo.wh = F(); Wo.ws_b = F(); Wo.ws_w = F(); Wo.wv = nullptr;
    // 'Wv': ln_b, ln_g, wh(3,16), ws_b(100), ws_w(22,100), wv(16,16)
    const float* Wv_ln_b = F(); const float* Wv_ln_g = F();
    GvpW Wv; Wv.wh = F(); Wv.ws_b = F(); Wv.ws_w = F(); Wv.wv = F();
    // 'gate': b1, b2, w1(100,16), w2(16,1)
    const float* g_b1 = F(); const float* g_b2 = F();
    const float* g_w1 = F(); const float* g_w2 = F();
    // 'layers' x3: ff[0], ff[1], msg[0..2] (each wh,ws_b,ws_w,wv), n0_b,n0_g,n1_b,n1_g
    struct Layer { GvpW ff0, ff1, m0, m1, m2; const float *n0_b,*n0_g,*n1_b,*n1_g; } L[3];
    for (int l = 0; l < 3; ++l) {
        L[l].ff0.wh=F(); L[l].ff0.ws_b=F(); L[l].ff0.ws_w=F(); L[l].ff0.wv=F();
        L[l].ff1.wh=F(); L[l].ff1.ws_b=F(); L[l].ff1.ws_w=F(); L[l].ff1.wv=F();
        L[l].m0.wh=F();  L[l].m0.ws_b=F();  L[l].m0.ws_w=F();  L[l].m0.wv=F();
        L[l].m1.wh=F();  L[l].m1.ws_b=F();  L[l].m1.ws_w=F();  L[l].m1.wv=F();
        L[l].m2.wh=F();  L[l].m2.ws_b=F();  L[l].m2.ws_w=F();  L[l].m2.wv=F();
        L[l].n0_b=F(); L[l].n0_g=F(); L[l].n1_b=F(); L[l].n1_g=F();
    }
    // 'mlp': b0,b1,b2,r0_b1,r0_b2,r0_w1,r0_w2,r1_b1,r1_b2,r1_w1,r1_w2,w0,w1,w2
    const float* m_b0=F();   const float* m_b1=F();   const float* m_b2=F();
    const float* m_r0b1=F(); const float* m_r0b2=F(); const float* m_r0w1=F(); const float* m_r0w2=F();
    const float* m_r1b1=F(); const float* m_r1b2=F(); const float* m_r1w1=F(); const float* m_r1w2=F();
    const float* m_w0=F();   const float* m_w1=F();   const float* m_w2=F();
    // data inputs
    const float* h_V_s = F();
    const float* h_V_v = F();
    const float* h_E_s = F();
    const float* h_E_v = F();
    const int* edge_index = (const int*)d_in[ci++];
    const int* water      = (const int*)d_in[ci++];
    const int* batch      = (const int*)d_in[ci++];
    const int* src = edge_index;
    const int* dst = edge_index + NE;

    // ---- workspace bump allocator ----
    unsigned char* wsBase = (unsigned char*)d_ws;
    size_t wsOff = 0;
    auto alloc = [&](size_t bytes) -> void* {
        wsOff = (wsOff + 255) & ~(size_t)255;
        void* p = wsBase + wsOff; wsOff += bytes; return p;
    };
    f16*   Abig  = (f16*)alloc((size_t)NE*288*sizeof(f16));   // staged A (max Kp=288 edge / 448 node)
    f16*   Bb    = (f16*)alloc((size_t)448*400*sizeof(f16));  // packed transposed weight [Np,Kp]
    float* biasP = (float*)alloc((size_t)400*sizeof(float));  // zero-padded bias [Np]
    float* CsE   = (float*)alloc((size_t)NE*LDN*sizeof(float)); // edge C / node FF C (N*400 fits)
    float* esE   = (float*)alloc((size_t)NE*32*sizeof(float));
    float* sTmpE = (float*)alloc((size_t)NE*32*sizeof(float));
    float* eVE   = (float*)alloc((size_t)NE*3*sizeof(float));
    float* eVt   = (float*)alloc((size_t)NE*3*sizeof(float));
    float* VoE1  = (float*)alloc((size_t)NE*48*sizeof(float));
    float* VoE2  = (float*)alloc((size_t)NE*48*sizeof(float));
    float* sA    = (float*)alloc((size_t)NN*LDN*sizeof(float));
    float* sB    = (float*)alloc((size_t)NN*LDN*sizeof(float));
    float* VA    = (float*)alloc((size_t)NN*48*sizeof(float));
    float* VB    = (float*)alloc((size_t)NN*48*sizeof(float));
    float* s1b   = (float*)alloc((size_t)NN*LDN*sizeof(float));
    float* V1b   = (float*)alloc((size_t)NN*48*sizeof(float));
    float* VoF   = (float*)alloc((size_t)NN*96*sizeof(float));
    float* Vff   = (float*)alloc((size_t)NN*48*sizeof(float));
    float* dsN   = (float*)alloc((size_t)NN*LDN*sizeof(float));
    float* dVN   = (float*)alloc((size_t)NN*48*sizeof(float));
    float* Cff   = (float*)alloc((size_t)NN*LDN*sizeof(float));
    float* outN  = (float*)alloc((size_t)NN*LDN*sizeof(float));
    float* cntN  = (float*)alloc((size_t)NN*sizeof(float));
    float* cntG  = (float*)alloc((size_t)NG*sizeof(float));
    float* gcB   = (float*)alloc((size_t)NG*LDN*sizeof(float));
    float* rep   = (float*)alloc((size_t)NG*LDN*sizeof(float));
    float* gBuf  = (float*)alloc((size_t)NN*sizeof(float));

    auto packW = [&](const float* W, const float* bias, int K, int N, int Kp, int Np) {
        int tot = Kp*Np;
        k_pack_wT<<<dim3((tot + 255)/256), dim3(256), 0, stream>>>(W, K, N, Bb, Kp, Np);
        k_pack_bias<<<dim3((Np + 255)/256), dim3(256), 0, stream>>>(bias, N, biasP, Np);
    };
    auto gemm = [&](const f16* A, int Kp, int Np, float* C, int ldc, int rows, int act) {
        k_gemm_wmma<<<dim3(rows/16, Np/16), dim3(32), 0, stream>>>(
            A, Kp, Bb, Kp, biasP, C, ldc, Kp/32, act);
    };
    auto zero = [&](float* p, long n) {
        k_zero<<<dim3(2048), dim3(256), 0, stream>>>(p, n);
    };

    const int TB = 256;
    const int gbN = (NN + TB - 1)/TB, gbE = (NE + TB - 1)/TB;

    // ---- edge counts per dst node (same for all layers) ----
    zero(cntN, NN);
    k_count<<<dim3(gbE), dim3(TB), 0, stream>>>(dst, NE, cntN);

    // ---- edge embedding: LN + GVP(We) ----
    k_ln_res<<<dim3(gbE), dim3(TB), 0, stream>>>(h_E_s, 32, nullptr, 0, 32,
        h_E_v, nullptr, 1, We_ln_g, We_ln_b, sTmpE, 32, eVt, NE);
    k_gvp_vec<<<dim3(gbE), dim3(TB), 0, stream>>>(sTmpE, 32, 32, eVt, 1,
        We.wh, 1, We.wv, 1, 0, eVE, Abig, 64, NE);
    packW(We.ws_w, We.ws_b, 33, 32, 64, 32);
    gemm(Abig, 64, 32, esE, 32, NE, 0);

    // ---- node embedding: LN + GVP(Wv) ----
    k_ln_res<<<dim3(gbN), dim3(TB), 0, stream>>>(h_V_s, 6, nullptr, 0, 6,
        h_V_v, nullptr, 3, Wv_ln_g, Wv_ln_b, s1b, LDN, V1b, NN);
    k_gvp_vec<<<dim3(gbN), dim3(TB), 0, stream>>>(s1b, LDN, 6, V1b, 3,
        Wv.wh, 16, Wv.wv, 16, 0, VA, Abig, 32, NN);
    packW(Wv.ws_w, Wv.ws_b, 22, 100, 32, LDN);
    gemm(Abig, 32, LDN, sA, LDN, NN, 0);

    float* curS = sA; float* curV = VA; float* altS = sB; float* altV = VB;

    // ---- 3 conv layers ----
    for (int l = 0; l < 3; ++l) {
        const Layer& P = L[l];
        // message GVP 0 (gather + vec track), then WMMA GEMM + relu
        k_edge_msg0<<<dim3(gbE), dim3(TB), 0, stream>>>(curS, LDN, curV, esE, 32,
            eVE, src, dst, P.m0.wh, P.m0.wv, VoE1, Abig, 288);
        packW(P.m0.ws_w, P.m0.ws_b, 265, 100, 288, LDN);
        gemm(Abig, 288, LDN, CsE, LDN, NE, 1);
        // message GVP 1
        k_gvp_vec<<<dim3(gbE), dim3(TB), 0, stream>>>(CsE, LDN, 100, VoE1, 16,
            P.m1.wh, 16, P.m1.wv, 16, 1, VoE2, Abig, 128, NE);
        packW(P.m1.ws_w, P.m1.ws_b, 116, 100, 128, LDN);
        gemm(Abig, 128, LDN, CsE, LDN, NE, 1);
        // message GVP 2 (no activation, no gate)
        k_gvp_vec<<<dim3(gbE), dim3(TB), 0, stream>>>(CsE, LDN, 100, VoE2, 16,
            P.m2.wh, 16, P.m2.wv, 16, 0, VoE1, Abig, 128, NE);
        packW(P.m2.ws_w, P.m2.ws_b, 116, 100, 128, LDN);
        gemm(Abig, 128, LDN, CsE, LDN, NE, 0);
        // segment mean over dst
        zero(dsN, (long)NN*LDN); zero(dVN, (long)NN*48);
        k_scatter_add<<<dim3((int)(((long)NE*100 + TB-1)/TB)), dim3(TB), 0, stream>>>(
            CsE, LDN, 100, dst, NE, dsN, LDN);
        k_scatter_add<<<dim3((int)(((long)NE*48 + TB-1)/TB)), dim3(TB), 0, stream>>>(
            VoE1, 48, 48, dst, NE, dVN, 48);
        k_div_cnt<<<dim3((int)(((long)NN*100 + TB-1)/TB)), dim3(TB), 0, stream>>>(
            dsN, LDN, 100, cntN, NN);
        k_div_cnt<<<dim3((int)(((long)NN*48 + TB-1)/TB)), dim3(TB), 0, stream>>>(
            dVN, 48, 48, cntN, NN);
        // s1 = LN(s + ds), V1 = (V + dV)/vn
        k_ln_res<<<dim3(gbN), dim3(TB), 0, stream>>>(curS, LDN, dsN, LDN, 100,
            curV, dVN, 16, P.n0_g, P.n0_b, s1b, LDN, V1b, NN);
        // FF GVP 0: 100/16 -> 400/32 (relu, sigmoid gate)
        k_gvp_vec<<<dim3(gbN), dim3(TB), 0, stream>>>(s1b, LDN, 100, V1b, 16,
            P.ff0.wh, 32, P.ff0.wv, 32, 1, VoF, Abig, 160, NN);
        packW(P.ff0.ws_w, P.ff0.ws_b, 132, 400, 160, 400);
        gemm(Abig, 160, 400, CsE, 400, NN, 1);
        // FF GVP 1: 400/32 -> 100/16 (no act)
        k_gvp_vec<<<dim3(gbN), dim3(TB), 0, stream>>>(CsE, 400, 400, VoF, 32,
            P.ff1.wh, 32, P.ff1.wv, 16, 0, Vff, Abig, 448, NN);
        packW(P.ff1.ws_w, P.ff1.ws_b, 432, 100, 448, LDN);
        gemm(Abig, 448, LDN, Cff, LDN, NN, 0);
        // s2 = LN(s1 + ff), then water_shell mask select vs layer input
        k_ln_res<<<dim3(gbN), dim3(TB), 0, stream>>>(s1b, LDN, Cff, LDN, 100,
            V1b, Vff, 16, P.n1_g, P.n1_b, altS, LDN, altV, NN);
        k_apply_mask<<<dim3(gbN), dim3(TB), 0, stream>>>(altS, LDN, curS, LDN,
            altV, curV, water, NN);
        float* t;
        t = curS; curS = altS; altS = t;
        t = curV; curV = altV; altV = t;
    }

    // ---- output GVP: LN + (100/16 -> 100 scalar, relu) ----
    k_ln_res<<<dim3(gbN), dim3(TB), 0, stream>>>(curS, LDN, nullptr, 0, 100,
        curV, nullptr, 16, Wo_ln_g, Wo_ln_b, s1b, LDN, V1b, NN);
    k_gvp_vec<<<dim3(gbN), dim3(TB), 0, stream>>>(s1b, LDN, 100, V1b, 16,
        Wo.wh, 16, nullptr, 0, 0, nullptr, Abig, 128, NN);
    packW(Wo.ws_w, Wo.ws_b, 116, 100, 128, LDN);
    gemm(Abig, 128, LDN, outN, LDN, NN, 1);

    // ---- global context: out += 0.5 * seg_mean(out, batch)[batch] ----
    zero(gcB, (long)NG*LDN); zero(cntG, NG);
    k_count<<<dim3(gbN), dim3(TB), 0, stream>>>(batch, NN, cntG);
    k_scatter_add<<<dim3((int)(((long)NN*100 + TB-1)/TB)), dim3(TB), 0, stream>>>(
        outN, LDN, 100, batch, NN, gcB, LDN);
    k_div_cnt<<<dim3((NG*100 + TB-1)/TB), dim3(TB), 0, stream>>>(gcB, LDN, 100, cntG, NG);
    k_add_gc<<<dim3((int)(((long)NN*100 + TB-1)/TB)), dim3(TB), 0, stream>>>(
        outN, LDN, gcB, LDN, batch, NN);

    // ---- attention pooling + head MLP ----
    k_gate<<<dim3(gbN), dim3(TB), 0, stream>>>(outN, LDN, g_w1, g_b1, g_w2, g_b2, gBuf, NN);
    k_pool<<<dim3(NG), dim3(256), 0, stream>>>(outN, LDN, gBuf, batch, rep, LDN);
    k_mlp<<<dim3(1), dim3(32), 0, stream>>>(rep, LDN,
        m_w0, m_b0, m_r0w1, m_r0b1, m_r0w2, m_r0b2,
        m_w1, m_b1, m_r1w1, m_r1b1, m_r1w2, m_r1b2,
        m_w2, m_b2, (float*)d_out);
}
